// Block_9122510536840
// MI455X (gfx1250) — matrix operand order repaired and checked by reading, not compiled, not run on hardware
//
#include <hip/hip_runtime.h>
#include <hip/hip_bf16.h>

// ---------------------------------------------------------------------------
// Types for CDNA5 WMMA (wave32): v_wmma_f32_16x16x32_bf16
// ---------------------------------------------------------------------------
typedef __attribute__((ext_vector_type(16))) __bf16 v16bf;
typedef __attribute__((ext_vector_type(8)))  float  v8f;

__device__ __forceinline__ unsigned short f2bf(float x) {
    unsigned int u = __float_as_uint(x);
    unsigned int r = (u + 0x7FFFu + ((u >> 16) & 1u)) >> 16;   // RNE
    return (unsigned short)r;
}

__device__ __forceinline__ float nan2num(float x) {
    if (isnan(x)) return 0.f;
    if (isinf(x)) return x > 0.f ? 3.402823466e38f : -3.402823466e38f;
    return x;
}

__device__ __forceinline__ v16bf ld_frag(const unsigned short* p0,
                                         const unsigned short* p1) {
    union { uint4 q[2]; v16bf v; } u;
    u.q[0] = *reinterpret_cast<const uint4*>(p0);   // ds_load_b128
    u.q[1] = *reinterpret_cast<const uint4*>(p1);
    return u.v;
}

// ---------------------------------------------------------------------------
// GEMM:  C[M,N] = A[M,K] @ Wt^T + bias,  Wt stored bf16 [N,K]
// A either: (a) fused gather-concat of up to 10 fp32 chunks (128 cols each),
//           (b) direct bf16 [M,K] (hidden activations).
// Epilogue: outH!=null -> LeakyReLU + bf16 store; else fp32 + bias store.
// ---------------------------------------------------------------------------
#define TM 128
#define TN 128
#define TK 64
#define LDP 72   // padded LDS pitch (ushorts): 144B rows, 16B aligned, bank-clean

struct ADesc {
    const float*     ptr[10];
    const long long* gidx[10];
    int              ld[10];
};

__global__ __launch_bounds__(256)
void gemm_bf16_wmma(ADesc ad,
                    const unsigned short* __restrict__ aH,
                    const unsigned short* __restrict__ Wt,
                    const float* __restrict__ bias,
                    float* __restrict__ outF,
                    unsigned short* __restrict__ outH,
                    int M, int N, int K)
{
    __shared__ unsigned short Al[TM * LDP];
    __shared__ unsigned short Bl[TN * LDP];

    const int t    = threadIdx.x;
    const int lane = t & 31, wid = t >> 5;
    const int wm   = (wid & 1) * 64;     // 2 waves along M
    const int wn   = (wid >> 1) * 32;    // 4 waves along N
    const int g16  = lane >> 4;          // lane half (ISA fragment layout)
    const int lm   = lane & 15;

    const int rowL = t >> 1;             // 0..127 tile row (A) / n row (B)
    const int khL  = (t & 1) * 32;       // k half within 64-wide K tile
    const long long rowG = (long long)blockIdx.x * TM + rowL;
    const long long rc   = rowG < M ? rowG : (long long)(M - 1);
    const int nG   = blockIdx.y * TN + rowL;

    v8f acc[4][2];
#pragma unroll
    for (int i = 0; i < 4; ++i)
#pragma unroll
        for (int j = 0; j < 2; ++j) acc[i][j] = (v8f)0.f;

    for (int k0 = 0; k0 < K; k0 += TK) {
        // ---- stage A tile (128 x 64 bf16) ----
        union { unsigned short s[32]; uint4 q[4]; } abuf;
        if (aH) {
            const unsigned short* p = aH + (size_t)rc * K + k0 + khL;
            const uint4* p4 = (const uint4*)p;
#pragma unroll
            for (int i = 0; i < 4; ++i) abuf.q[i] = p4[i];
            if (k0 + TK < K) __builtin_prefetch(p + TK, 0, 3);   // global_prefetch_b8
        } else {
            const int kg = k0 + khL;                // chunk-aligned (32 | 128)
            const int c  = kg >> 7;
            const long long ri = ad.gidx[c] ? ad.gidx[c][rc] : rc;
            const float* p = ad.ptr[c] + (size_t)ri * ad.ld[c] + (kg & 127);
            const float4* p4 = (const float4*)p;
#pragma unroll
            for (int i = 0; i < 8; ++i) {
                float4 f = p4[i];
                abuf.s[4*i+0] = f2bf(f.x); abuf.s[4*i+1] = f2bf(f.y);
                abuf.s[4*i+2] = f2bf(f.z); abuf.s[4*i+3] = f2bf(f.w);
            }
        }
        {
            uint4* dst = (uint4*)&Al[rowL * LDP + khL];
#pragma unroll
            for (int i = 0; i < 4; ++i) dst[i] = abuf.q[i];
        }
        // ---- stage B tile: Wt bf16 [N,K] rows -> straight b128 copies ----
        {
            const unsigned short* p = Wt + (size_t)nG * K + k0 + khL;
            const uint4* p4 = (const uint4*)p;
            uint4* dst = (uint4*)&Bl[rowL * LDP + khL];
#pragma unroll
            for (int i = 0; i < 4; ++i) dst[i] = p4[i];
            if (k0 + TK < K) __builtin_prefetch(p + TK, 0, 3);
        }
        __syncthreads();

#pragma unroll
        for (int ks = 0; ks < TK; ks += 32) {
            v16bf af[4], bfr[2];
#pragma unroll
            for (int mi = 0; mi < 4; ++mi) {
                const unsigned short* b = &Al[(wm + mi*16 + lm) * LDP + ks];
                // A 16x32 bf16 layout: lane half g: K = 8g..8g+7, 16+8g..16+8g+7
                af[mi] = ld_frag(b + 8*g16, b + 16 + 8*g16);
            }
#pragma unroll
            for (int ni = 0; ni < 2; ++ni) {
                // B 32x16 bf16 layout: lane half g: K = 8g..8g+7, 16+8g..16+8g+7, N = lane%16
                const unsigned short* b = &Bl[(wn + ni*16 + lm) * LDP + ks];
                bfr[ni] = ld_frag(b + 8*g16, b + 16 + 8*g16);
            }
#pragma unroll
            for (int mi = 0; mi < 4; ++mi)
#pragma unroll
                for (int ni = 0; ni < 2; ++ni)
                    acc[mi][ni] = __builtin_amdgcn_wmma_f32_16x16x32_bf16(
                        false, af[mi], false, bfr[ni],
                        (short)0, acc[mi][ni], false, false);
        }
        __syncthreads();
    }

    // ---- epilogue (C layout: m = 8*g16 + j, n = lane%16) ----
#pragma unroll
    for (int mi = 0; mi < 4; ++mi) {
#pragma unroll
        for (int ni = 0; ni < 2; ++ni) {
            const int col = blockIdx.y * TN + wn + ni*16 + lm;
            const float bv = bias[col];
#pragma unroll
            for (int j = 0; j < 8; ++j) {
                const long long row =
                    (long long)blockIdx.x * TM + wm + mi*16 + g16*8 + j;
                if (row < M) {
                    float v = acc[mi][ni][j] + bv;
                    if (outH) {
                        v = v >= 0.f ? v : 0.01f * v;           // LeakyReLU
                        outH[(size_t)row * N + col] = f2bf(v);
                    } else {
                        outF[(size_t)row * N + col] = v;
                    }
                }
            }
        }
    }
}

// ---------------------------------------------------------------------------
// Weight prep: fp32 [K,N] -> bf16 [N,K] (transposed for contiguous B staging)
// ---------------------------------------------------------------------------
__global__ void k_wprep(const float* __restrict__ W,
                        unsigned short* __restrict__ Wt, int K, int N) {
    int i = blockIdx.x * blockDim.x + threadIdx.x;
    if (i >= K * N) return;
    int k = i / N, n = i - k * N;
    Wt[(size_t)n * K + k] = f2bf(W[i]);
}

// ---------------------------------------------------------------------------
// Scatter / stats kernels (C = 256 columns, hardware fp32 atomics)
// ---------------------------------------------------------------------------
__global__ void k_scatter1(const float* __restrict__ msg,
                           const long long* __restrict__ src,
                           float* __restrict__ sum, float* __restrict__ sumsq,
                           float* __restrict__ cnt, long long E) {
    long long tid = (long long)blockIdx.x * blockDim.x + threadIdx.x;
    if (tid >= (E << 8)) return;
    long long e = tid >> 8; int c = (int)(tid & 255);
    int s = (int)src[e];
    float v = msg[tid];
    unsafeAtomicAdd(&sum[(size_t)s * 256 + c], v);
    unsafeAtomicAdd(&sumsq[(size_t)s * 256 + c], v * v);
    if (c == 0) unsafeAtomicAdd(&cnt[s], 1.0f);
}

__global__ void k_nstats1(float* __restrict__ sum, float* __restrict__ sumsq,
                          const float* __restrict__ cnt, long long NS) {
    long long tid = (long long)blockIdx.x * blockDim.x + threadIdx.x;
    if (tid >= (NS << 8)) return;
    long long n = tid >> 8;
    float cm   = fmaxf(cnt[n], 1.0f);
    float mean = nan2num(sum[tid] / cm);
    float var  = sumsq[tid] / cm - mean * mean;
    var = var >= 0.f ? var : 0.01f * var;          // LeakyReLU per reference
    var = nan2num(var);
    sum[tid]   = mean;                              // in-place: sum -> mean
    sumsq[tid] = sqrtf(var + 1e-6f);                // in-place: sumsq -> std
}

__global__ void k_scatter2(const float* __restrict__ msg,
                           const long long* __restrict__ src,
                           const float* __restrict__ mean,
                           float* __restrict__ s3, float* __restrict__ s4,
                           long long E) {
    long long tid = (long long)blockIdx.x * blockDim.x + threadIdx.x;
    if (tid >= (E << 8)) return;
    long long e = tid >> 8; int c = (int)(tid & 255);
    int s = (int)src[e];
    float d = msg[tid] - mean[(size_t)s * 256 + c];
    float d3 = d * d * d;
    unsafeAtomicAdd(&s3[(size_t)s * 256 + c], d3);
    unsafeAtomicAdd(&s4[(size_t)s * 256 + c], d3 * d);
}

__global__ void k_nstats2(float* __restrict__ s3, float* __restrict__ s4,
                          const float* __restrict__ stdb,
                          const float* __restrict__ cnt, long long NS) {
    long long tid = (long long)blockIdx.x * blockDim.x + threadIdx.x;
    if (tid >= (NS << 8)) return;
    long long n = tid >> 8;
    float cm = fmaxf(cnt[n], 1.0f);
    float sd = stdb[tid];
    float s3m = s3[tid] / cm, s4m = s4[tid] / cm;
    s3[tid] = nan2num(s3m / (sd * sd * sd));        // skew
    s4[tid] = nan2num(s4m / (sd * sd * sd * sd));   // kurt
}

__global__ void k_scatsum(const float* __restrict__ msg,
                          const long long* __restrict__ tgt,
                          float* __restrict__ agg, long long E) {
    long long tid = (long long)blockIdx.x * blockDim.x + threadIdx.x;
    if (tid >= (E << 8)) return;
    long long e = tid >> 8; int c = (int)(tid & 255);
    int s = (int)tgt[e];
    unsafeAtomicAdd(&agg[(size_t)s * 256 + c], msg[tid]);
}

// ---------------------------------------------------------------------------
// BatchNorm (training-mode batch stats) + column sums
// ---------------------------------------------------------------------------
__global__ void k_bnstats(const float* __restrict__ x,
                          float* __restrict__ stats, long long M) {
    int c = threadIdx.x;            // 128 threads
    float s = 0.f, s2 = 0.f;
    for (long long r = blockIdx.x; r < M; r += gridDim.x) {
        float v = x[r * 128 + c];
        s += v; s2 += v * v;
    }
    unsafeAtomicAdd(&stats[c], s);
    unsafeAtomicAdd(&stats[128 + c], s2);
}

__global__ void k_bnapply(float* __restrict__ x, const float* __restrict__ stats,
                          const float* __restrict__ g, const float* __restrict__ b,
                          long long M) {
    long long tid = (long long)blockIdx.x * blockDim.x + threadIdx.x;
    if (tid >= M * 128) return;
    int c = (int)(tid & 127);
    float m   = stats[c] / (float)M;
    float var = stats[128 + c] / (float)M - m * m;
    x[tid] = g[c] * (x[tid] - m) * rsqrtf(var + 1e-5f) + b[c];
}

__global__ void k_colsum(const float* __restrict__ x, float* __restrict__ o,
                         long long M) {
    int c = threadIdx.x;
    float s = 0.f;
    for (long long r = blockIdx.x; r < M; r += gridDim.x) s += x[r * 128 + c];
    unsafeAtomicAdd(&o[c], s);
}

// ---------------------------------------------------------------------------
// Global model: [1,384] MLP + RMSNorm, one workgroup
// ---------------------------------------------------------------------------
__global__ void k_global(const float* __restrict__ xu,
                         const float* __restrict__ cs, const float* __restrict__ ct,
                         const float* __restrict__ W1, const float* __restrict__ b1,
                         const float* __restrict__ W2, const float* __restrict__ b2,
                         const float* __restrict__ gg, float* __restrict__ out,
                         float invNS, float invNT) {
    __shared__ float h[384], h2[384], gv[128], sq[128], red;
    int t = threadIdx.x;            // 384 threads
    if      (t < 128) h[t] = xu[t];
    else if (t < 256) h[t] = cs[t - 128] * invNS;
    else              h[t] = ct[t - 256] * invNT;
    __syncthreads();
    float a = b1[t];
    for (int i = 0; i < 384; ++i) a += h[i] * W1[i * 384 + t];
    h2[t] = a >= 0.f ? a : 0.01f * a;
    __syncthreads();
    if (t < 128) {
        float g = b2[t];
        for (int i = 0; i < 384; ++i) g += h2[i] * W2[i * 128 + t];
        gv[t] = g; sq[t] = g * g;
    }
    __syncthreads();
    if (t == 0) { float s = 0.f; for (int i = 0; i < 128; ++i) s += sq[i]; red = s; }
    __syncthreads();
    if (t < 128)
        out[t] = gv[t] * rsqrtf(red / 128.f + 1.1920929e-7f) * gg[t];
}

// ---------------------------------------------------------------------------
// Host orchestration
// ---------------------------------------------------------------------------
static void launch_gemm(hipStream_t s, const ADesc& ad, const unsigned short* aH,
                        const unsigned short* Wt, const float* bias,
                        float* outF, unsigned short* outH,
                        long long M, int N, int K) {
    dim3 g((unsigned)((M + TM - 1) / TM), (unsigned)(N / TN));
    gemm_bf16_wmma<<<g, 256, 0, s>>>(ad, aH, Wt, bias, outF, outH, (int)M, N, K);
}

extern "C" void kernel_launch(void* const* d_in, const int* in_sizes, int n_in,
                              void* d_out, int out_size, void* d_ws, size_t ws_size,
                              hipStream_t stream) {
    const int D = 128;
    const float*     x_s   = (const float*)d_in[0];
    const float*     x_t   = (const float*)d_in[1];
    const long long* ei    = (const long long*)d_in[2];
    const float*     eattr = (const float*)d_in[3];
    const float*     x_u   = (const float*)d_in[4];
    const float* P[31];
    for (int i = 0; i < 31; ++i) P[i] = (const float*)d_in[5 + i];
    const float *edge_W1=P[0],*edge_b1=P[1],*edge_W2=P[2],*edge_b2=P[3],*edge_g=P[4],*edge_be=P[5];
    const float *sm_W1=P[6],*sm_b1=P[7],*sm_W2=P[8],*sm_b2=P[9];
    const float *su_W1=P[10],*su_b1=P[11],*su_W2=P[12],*su_b2=P[13],*src_g=P[14],*src_be=P[15];
    const float *tm_W1=P[16],*tm_b1=P[17],*tm_W2=P[18],*tm_b2=P[19];
    const float *tu_W1=P[20],*tu_b1=P[21],*tu_W2=P[22],*tu_b2=P[23],*tgt_g=P[24],*tgt_be=P[25];
    const float *g_W1=P[26],*g_b1=P[27],*g_W2=P[28],*g_b2=P[29],*g_g=P[30];

    const long long NS = in_sizes[0] / D;
    const long long NT = in_sizes[1] / D;
    const long long E  = in_sizes[3] / D;
    const long long* srcI = ei;
    const long long* tgtI = ei + E;

    // ---- output layout (tuple order: x_s, x_t, edge_index, edge_attr, x_u) ----
    float* out = (float*)d_out;
    size_t o_xt = (size_t)NS * D;
    size_t o_ei = o_xt + (size_t)NT * D;
    size_t tail = (size_t)E * D + D;
    size_t ei_slots = (size_t)out_size - o_ei - tail;   // edge_index passthrough slots
    size_t o_ea = o_ei + ei_slots;
    size_t o_xu = o_ea + (size_t)E * D;
    float* xs_o = out;
    float* xt_o = out + o_xt;
    float* ea_o = out + o_ea;
    float* xu_o = out + o_xu;

    // ---- workspace bump allocator ----
    char* ws = (char*)d_ws;
    size_t off = 0;
    auto alloc = [&](size_t bytes) -> void* {
        off = (off + 255) & ~(size_t)255;
        void* p = ws + off;
        off += bytes;
        return p;
    };
    struct { const float* W; int K, N; } wspec[10] = {
        {edge_W1, 4*D, 4*D}, {edge_W2, 4*D, D},
        {sm_W1, 2*D, 2*D},   {sm_W2, 2*D, 2*D},
        {su_W1, 10*D, 10*D}, {su_W2, 10*D, D},
        {tm_W1, 2*D, 2*D},   {tm_W2, 2*D, 2*D},
        {tu_W1, 4*D, 4*D},   {tu_W2, 4*D, D}};
    unsigned short* wt[10];
    for (int i = 0; i < 10; ++i)
        wt[i] = (unsigned short*)alloc((size_t)wspec[i].K * wspec[i].N * 2);
    unsigned short* hid  = (unsigned short*)alloc((size_t)E * 512 * 2);  // max hidden
    float* msg  = (float*)alloc((size_t)E * 256 * 4);
    float* sumA = (float*)alloc((size_t)NS * 256 * 4);   // -> mean
    float* sumB = (float*)alloc((size_t)NS * 256 * 4);   // -> std
    float* sumC = (float*)alloc((size_t)NS * 256 * 4);   // -> skew
    float* sumD = (float*)alloc((size_t)NS * 256 * 4);   // -> kurt
    float* cnt  = (float*)alloc((size_t)NS * 4);
    float* agg  = (float*)alloc((size_t)NT * 256 * 4);
    float* bn1  = (float*)alloc(256 * 4);
    float* bn2  = (float*)alloc(256 * 4);
    float* bn3  = (float*)alloc(256 * 4);
    float* cs_s = (float*)alloc(128 * 4);
    float* cs_t = (float*)alloc(128 * 4);

    // ---- zero accumulators (every call; no cross-call state) ----
    hipMemsetAsync(sumA, 0, (size_t)NS * 256 * 4, stream);
    hipMemsetAsync(sumB, 0, (size_t)NS * 256 * 4, stream);
    hipMemsetAsync(sumC, 0, (size_t)NS * 256 * 4, stream);
    hipMemsetAsync(sumD, 0, (size_t)NS * 256 * 4, stream);
    hipMemsetAsync(cnt,  0, (size_t)NS * 4, stream);
    hipMemsetAsync(agg,  0, (size_t)NT * 256 * 4, stream);
    hipMemsetAsync(bn1,  0, 256 * 4, stream);
    hipMemsetAsync(bn2,  0, 256 * 4, stream);
    hipMemsetAsync(bn3,  0, 256 * 4, stream);
    hipMemsetAsync(cs_s, 0, 128 * 4, stream);
    hipMemsetAsync(cs_t, 0, 128 * 4, stream);

    // ---- weight conversion fp32[K,N] -> bf16[N,K] ----
    for (int i = 0; i < 10; ++i) {
        int tot = wspec[i].K * wspec[i].N;
        k_wprep<<<(tot + 255) / 256, 256, 0, stream>>>(wspec[i].W, wt[i],
                                                       wspec[i].K, wspec[i].N);
    }

    ADesc z = {};   // zeroed template
    const long long EC  = E << 8;   // E*256 scatter threads
    const long long NSC = NS << 8;

    // ================= 1) Edge MLP: [x_s[src], x_t[tgt], e, u] K=512 ==========
    {
        ADesc ad = z;
        ad.ptr[0] = x_s;   ad.gidx[0] = srcI; ad.ld[0] = 128;
        ad.ptr[1] = x_t;   ad.gidx[1] = tgtI; ad.ld[1] = 128;
        ad.ptr[2] = eattr; ad.gidx[2] = 0;    ad.ld[2] = 128;
        ad.ptr[3] = x_u;   ad.gidx[3] = 0;    ad.ld[3] = 0;    // broadcast
        launch_gemm(stream, ad, 0, wt[0], edge_b1, 0, hid, E, 512, 512);
        launch_gemm(stream, z, hid, wt[1], edge_b2, ea_o, 0, E, 128, 512);
        k_bnstats<<<512, 128, 0, stream>>>(ea_o, bn1, E);
        k_bnapply<<<(unsigned)((E * 128 + 255) / 256), 256, 0, stream>>>(
            ea_o, bn1, edge_g, edge_be, E);
    }
    // ================= 2) src message MLP: [x_t[tgt], e'] K=256 ===============
    {
        ADesc ad = z;
        ad.ptr[0] = x_t;  ad.gidx[0] = tgtI; ad.ld[0] = 128;
        ad.ptr[1] = ea_o; ad.gidx[1] = 0;    ad.ld[1] = 128;
        launch_gemm(stream, ad, 0, wt[2], sm_b1, 0, hid, E, 256, 256);
        launch_gemm(stream, z, hid, wt[3], sm_b2, msg, 0, E, 256, 256);
    }
    // ================= 3) scatter moments by src ==============================
    k_scatter1<<<(unsigned)((EC + 255) / 256), 256, 0, stream>>>(msg, srcI, sumA, sumB, cnt, E);
    k_nstats1<<<(unsigned)((NSC + 255) / 256), 256, 0, stream>>>(sumA, sumB, cnt, NS);
    k_scatter2<<<(unsigned)((EC + 255) / 256), 256, 0, stream>>>(msg, srcI, sumA, sumC, sumD, E);
    k_nstats2<<<(unsigned)((NSC + 255) / 256), 256, 0, stream>>>(sumC, sumD, sumB, cnt, NS);
    // ================= 4) src update MLP: K=1280 (10 chunks) ==================
    {
        ADesc ad = z;
        ad.ptr[0] = x_s;        ad.ld[0] = 128;
        ad.ptr[1] = sumA;       ad.ld[1] = 256;   // mean lo
        ad.ptr[2] = sumA + 128; ad.ld[2] = 256;   // mean hi
        ad.ptr[3] = sumB;       ad.ld[3] = 256;   // std lo
        ad.ptr[4] = sumB + 128; ad.ld[4] = 256;
        ad.ptr[5] = sumC;       ad.ld[5] = 256;   // skew
        ad.ptr[6] = sumC + 128; ad.ld[6] = 256;
        ad.ptr[7] = sumD;       ad.ld[7] = 256;   // kurt
        ad.ptr[8] = sumD + 128; ad.ld[8] = 256;
        ad.ptr[9] = x_u;        ad.ld[9] = 0;
        launch_gemm(stream, ad, 0, wt[4], su_b1, 0, hid, NS, 1280, 1280);
        launch_gemm(stream, z, hid, wt[5], su_b2, xs_o, 0, NS, 128, 1280);
        k_bnstats<<<512, 128, 0, stream>>>(xs_o, bn2, NS);
        k_bnapply<<<(unsigned)((NS * 128 + 255) / 256), 256, 0, stream>>>(
            xs_o, bn2, src_g, src_be, NS);
    }
    // ================= 5) tgt message MLP: [x_s'[src], e'] K=256 ==============
    {
        ADesc ad = z;
        ad.ptr[0] = xs_o; ad.gidx[0] = srcI; ad.ld[0] = 128;
        ad.ptr[1] = ea_o; ad.gidx[1] = 0;    ad.ld[1] = 128;
        launch_gemm(stream, ad, 0, wt[6], tm_b1, 0, hid, E, 256, 256);
        launch_gemm(stream, z, hid, wt[7], tm_b2, msg, 0, E, 256, 256);
    }
    // ================= 6) segment_sum by tgt ==================================
    k_scatsum<<<(unsigned)((EC + 255) / 256), 256, 0, stream>>>(msg, tgtI, agg, E);
    // ================= 7) tgt update MLP: [x_t, agg, u] K=512 =================
    {
        ADesc ad = z;
        ad.ptr[0] = x_t;       ad.ld[0] = 128;
        ad.ptr[1] = agg;       ad.ld[1] = 256;
        ad.ptr[2] = agg + 128; ad.ld[2] = 256;
        ad.ptr[3] = x_u;       ad.ld[3] = 0;
        launch_gemm(stream, ad, 0, wt[8], tu_b1, 0, hid, NT, 512, 512);
        launch_gemm(stream, z, hid, wt[9], tu_b2, xt_o, 0, NT, 128, 512);
        k_bnstats<<<512, 128, 0, stream>>>(xt_o, bn3, NT);
        k_bnapply<<<(unsigned)((NT * 128 + 255) / 256), 256, 0, stream>>>(
            xt_o, bn3, tgt_g, tgt_be, NT);
    }
    // ================= 8) global model ========================================
    k_colsum<<<256, 128, 0, stream>>>(xs_o, cs_s, NS);
    k_colsum<<<256, 128, 0, stream>>>(xt_o, cs_t, NT);
    k_global<<<1, 384, 0, stream>>>(x_u, cs_s, cs_t, g_W1, g_b1, g_W2, g_b2,
                                    g_g, xu_o, 1.0f / (float)NS, 1.0f / (float)NT);
    // ================= 9) edge_index passthrough ==============================
    if (ei_slots > 0)
        hipMemcpyAsync(out + o_ei, d_in[2], ei_slots * sizeof(float),
                       hipMemcpyDeviceToDevice, stream);
    (void)n_in; (void)ws_size;
}
